// CollaborativeMultiheadAttention_21663815041723
// MI455X (gfx1250) — compile-verified, matrix operand order checked by reading, not run on hardware
//
#include <hip/hip_runtime.h>

// ---------------------------------------------------------------------------
// CDNA5 / gfx1250 fused multi-head attention (bf16 WMMA, fp32 accumulate)
// S=1024, B=8, E=1024, H=16, Dh=64  (wave32, v_wmma_f32_16x16x32_bf16)
//
// v2: register-blocked GEMMs (64-row M tiles) to raise WMMA:load ratio,
//     LDS-staged double-buffered K/V tiles in attention for 8x WGP reuse.
// ---------------------------------------------------------------------------

typedef __attribute__((ext_vector_type(8)))  __bf16 v8bf;
typedef __attribute__((ext_vector_type(16))) __bf16 v16bf;
typedef __attribute__((ext_vector_type(8)))  float  v8f;

#define S_LEN 1024
#define BSZ   8
#define EMB   1024
#define NH    16
#define DH    64
#define MROWS (S_LEN * BSZ)   // 8192

__device__ __forceinline__ v8bf ld8(const __bf16* p) { return *(const v8bf*)p; }

__device__ __forceinline__ v16bf mk16(v8bf a, v8bf b) {
    v16bf r;
#pragma unroll
    for (int i = 0; i < 8; ++i) { r[i] = a[i]; r[i + 8] = b[i]; }
    return r;
}

__device__ __forceinline__ v8f zero8() {
    v8f z;
#pragma unroll
    for (int i = 0; i < 8; ++i) z[i] = 0.0f;
    return z;
}

__device__ __forceinline__ v8f bwmma(v16bf a, v16bf b, v8f c) {
    // D = A(16x32 bf16) * B(32x16 bf16) + C(16x16 f32)
    return __builtin_amdgcn_wmma_f32_16x16x32_bf16(false, a, false, b,
                                                   (short)0, c, false, false);
}

__device__ __forceinline__ float redmax16(float v) {
    v = fmaxf(v, __shfl_xor(v, 1, 32));
    v = fmaxf(v, __shfl_xor(v, 2, 32));
    v = fmaxf(v, __shfl_xor(v, 4, 32));
    v = fmaxf(v, __shfl_xor(v, 8, 32));
    return v;
}
__device__ __forceinline__ float redsum16(float v) {
    v += __shfl_xor(v, 1, 32);
    v += __shfl_xor(v, 2, 32);
    v += __shfl_xor(v, 4, 32);
    v += __shfl_xor(v, 8, 32);
    return v;
}

// --------------------------- conversion kernels ----------------------------

__global__ __launch_bounds__(256) void cvt_f32_bf16_k(const float* __restrict__ src,
                                                      __bf16* __restrict__ dst) {
    int i = (blockIdx.x * 256 + threadIdx.x) * 4;
#pragma unroll
    for (int j = 0; j < 4; ++j) dst[i + j] = (__bf16)src[i + j];
}

// wT[n][k] = (bf16) w[k][n]   (1024 x 1024)
__global__ __launch_bounds__(256) void transpose_bf16_k(const float* __restrict__ w,
                                                        __bf16* __restrict__ wT) {
    int idx = blockIdx.x * 256 + threadIdx.x;   // over 1M elements
    int k = idx >> 10, n = idx & 1023;
    wT[(size_t)n * EMB + k] = (__bf16)w[idx];
}

// ------------------------------ q/k/v GEMM ---------------------------------
// One wave -> 64x16 output tile (4 M-subtiles) of q,k,v simultaneously:
// per k-step 4 A-frags + 3 B-frags feed 12 WMMAs (1.17 b128-loads/WMMA).
// q scaled by Dh^-0.5; v gets the x residual and is written TRANSPOSED as
// vt[(b*NH+h)*DH + d][s] so the PV GEMM / LDS staging read contiguous keys.
__global__ __launch_bounds__(256) void qkv_proj_k(const __bf16* __restrict__ xb,
                                                  const float*  __restrict__ x,
                                                  const __bf16* __restrict__ wqT,
                                                  const __bf16* __restrict__ wkT,
                                                  const __bf16* __restrict__ wvT,
                                                  __bf16* __restrict__ qb,
                                                  __bf16* __restrict__ kb,
                                                  __bf16* __restrict__ vt) {
    const int wid  = threadIdx.x >> 5;
    const int lane = threadIdx.x & 31;
    const int tile = blockIdx.x * 8 + wid;        // 0 .. 8191
    const int tmb = tile >> 6;                    // 128 M-blocks of 64 rows
    const int tn  = tile & 63;                    // 64 N-tiles of 16 cols
    const int col = lane & 15;
    const int hi  = lane >> 4;
    const int klo = hi * 8;

    const __bf16* ap[4];
#pragma unroll
    for (int i = 0; i < 4; ++i)
        ap[i] = xb + (size_t)(tmb * 64 + i * 16 + col) * EMB;
    const __bf16* bq = wqT + (size_t)(tn * 16 + col) * EMB;
    const __bf16* bk = wkT + (size_t)(tn * 16 + col) * EMB;
    const __bf16* bv = wvT + (size_t)(tn * 16 + col) * EMB;

    v8f aq[4], ak[4], av[4];
#pragma unroll
    for (int i = 0; i < 4; ++i) { aq[i] = zero8(); ak[i] = zero8(); av[i] = zero8(); }

    for (int kk = 0; kk < EMB; kk += 32) {
        v16bf fq = mk16(ld8(bq + kk + klo), ld8(bq + kk + klo + 16));
        v16bf fk = mk16(ld8(bk + kk + klo), ld8(bk + kk + klo + 16));
        v16bf fv = mk16(ld8(bv + kk + klo), ld8(bv + kk + klo + 16));
#pragma unroll
        for (int i = 0; i < 4; ++i) {
            v16bf af = mk16(ld8(ap[i] + kk + klo), ld8(ap[i] + kk + klo + 16));
            aq[i] = bwmma(af, fq, aq[i]);
            ak[i] = bwmma(af, fk, ak[i]);
            av[i] = bwmma(af, fv, av[i]);
        }
    }

    const int n = tn * 16 + col;
    const int h = n >> 6, d = n & 63;
#pragma unroll
    for (int i = 0; i < 4; ++i) {
#pragma unroll
        for (int r = 0; r < 8; ++r) {
            const int row = tmb * 64 + i * 16 + r + klo;   // row = s*BSZ + b
            qb[(size_t)row * EMB + n] = (__bf16)(aq[i][r] * 0.125f); // Dh^-0.5
            kb[(size_t)row * EMB + n] = (__bf16)ak[i][r];
            const float vv = av[i][r] + x[(size_t)row * EMB + n];    // residual
            const int b = row & 7, s = row >> 3;
            vt[(((size_t)(b * NH + h)) * DH + d) * S_LEN + s] = (__bf16)vv;
        }
    }
}

// --------------------------- flash attention -------------------------------
// Block = 8 waves, all on the SAME (b,h); each wave owns one 16-query tile.
// Per 32-key block, K (32x64) and V^T (64x32) tiles are cooperatively staged
// into double-buffered LDS (next tile's global loads issued before compute),
// giving 8x reuse per WGP. Online softmax per wave; prob tile round-trips
// through per-wave LDS to convert C-layout -> A-layout (ds_load_b128).
__global__ __launch_bounds__(256) void attn_k(const __bf16* __restrict__ qb,
                                              const __bf16* __restrict__ kb,
                                              const __bf16* __restrict__ vt,
                                              const float*  __restrict__ mask,
                                              __bf16* __restrict__ ctx) {
    __shared__ __bf16 ktile[2][32][64];   // [buf][key][d]
    __shared__ __bf16 vtile[2][64][32];   // [buf][d][key]
    __shared__ __bf16 ptile[8][16][32];   // per-wave prob tile

    const int wid  = threadIdx.x >> 5;
    const int lane = threadIdx.x & 31;
    const int t    = threadIdx.x;
    const int bh   = blockIdx.x >> 3;             // 128 (b,h) batches
    const int qt   = (blockIdx.x & 7) * 8 + wid;  // query tile 0..63
    const int b = bh >> 4, h = bh & 15;
    const int col = lane & 15;
    const int hi  = lane >> 4;
    const int klo = hi * 8;

    // cooperative K/V staging offsets (per thread: 2 x 16B)
    const int kkey  = t >> 3, kpart = (t & 7) * 8;
    const int vd    = t >> 2, vpart = (t & 3) * 8;
    const __bf16* ksrc = kb + (size_t)(kkey * BSZ + b) * EMB + h * DH + kpart;
    const __bf16* vsrc = vt + ((size_t)bh * DH + vd) * S_LEN + vpart;

    // Q A-fragments (hoisted, reused across all key blocks)
    const __bf16* qrow = qb + (size_t)((qt * 16 + col) * BSZ + b) * EMB + h * DH;
    const v16bf qa0 = mk16(ld8(qrow + klo),      ld8(qrow + klo + 16));
    const v16bf qa1 = mk16(ld8(qrow + 32 + klo), ld8(qrow + 32 + klo + 16));

    v8f o0 = zero8(), o1 = zero8(), o2 = zero8(), o3 = zero8();
    float m[8], l[8];
#pragma unroll
    for (int r = 0; r < 8; ++r) { m[r] = -3.0e38f; l[r] = 0.0f; }

    // stage key block 0
    *(v8bf*)&ktile[0][kkey][kpart] = *(const v8bf*)(ksrc);
    *(v8bf*)&vtile[0][vd][vpart]   = *(const v8bf*)(vsrc);
    __syncthreads();

    for (int kt = 0; kt < S_LEN / 32; ++kt) {
        const int cur = kt & 1;
        const int k0  = kt * 32;
        // issue next tile's global->LDS staging before compute (overlap)
        if (kt + 1 < S_LEN / 32) {
            const size_t goff = (size_t)32 * BSZ * EMB;   // +32 keys in kb
            *(v8bf*)&ktile[cur ^ 1][kkey][kpart] =
                *(const v8bf*)(ksrc + (size_t)(kt + 1) * goff);
            *(v8bf*)&vtile[cur ^ 1][vd][vpart] =
                *(const v8bf*)(vsrc + (kt + 1) * 32);
        }

        // ---- scores: two 16x16 tiles from LDS K (K-dim = DH = 64)
        const __bf16* kr0 = &ktile[cur][col][0];
        const __bf16* kr1 = &ktile[cur][16 + col][0];
        v8f s0 = zero8(), s1 = zero8();
        s0 = bwmma(qa0, mk16(*(const v8bf*)(kr0 + klo),
                             *(const v8bf*)(kr0 + klo + 16)), s0);
        s0 = bwmma(qa1, mk16(*(const v8bf*)(kr0 + 32 + klo),
                             *(const v8bf*)(kr0 + 32 + klo + 16)), s0);
        s1 = bwmma(qa0, mk16(*(const v8bf*)(kr1 + klo),
                             *(const v8bf*)(kr1 + klo + 16)), s1);
        s1 = bwmma(qa1, mk16(*(const v8bf*)(kr1 + 32 + klo),
                             *(const v8bf*)(kr1 + 32 + klo + 16)), s1);

        const float mk0 = mask[b * S_LEN + k0 + col];
        const float mk1 = mask[b * S_LEN + k0 + 16 + col];

        // ---- online softmax (row stats shared within each 16-lane group)
        float corr[8];
#pragma unroll
        for (int r = 0; r < 8; ++r) {
            const float e0 = s0[r] + mk0;
            const float e1 = s1[r] + mk1;
            const float tmax = redmax16(fmaxf(e0, e1));
            const float mn = fmaxf(m[r], tmax);
            const float c  = __expf(m[r] - mn);
            const float p0 = __expf(e0 - mn);
            const float p1 = __expf(e1 - mn);
            l[r] = l[r] * c + redsum16(p0 + p1);
            m[r] = mn;
            corr[r] = c;
            ptile[wid][r + klo][col]      = (__bf16)p0;
            ptile[wid][r + klo][col + 16] = (__bf16)p1;
        }
#pragma unroll
        for (int r = 0; r < 8; ++r) {
            o0[r] *= corr[r]; o1[r] *= corr[r];
            o2[r] *= corr[r]; o3[r] *= corr[r];
        }

        // ---- reload probs in A-fragment layout (same-wave DS in-order)
        const __bf16* prow = &ptile[wid][col][0];
        const v16bf pa = mk16(*(const v8bf*)(prow + klo),
                              *(const v8bf*)(prow + klo + 16));

        // ---- ctx += P(16x32) * V(32x64): 4 WMMAs from LDS V^T
        {
            const __bf16* vr = &vtile[cur][0 * 16 + col][0];
            o0 = bwmma(pa, mk16(*(const v8bf*)(vr + klo),
                                *(const v8bf*)(vr + klo + 16)), o0);
        }
        {
            const __bf16* vr = &vtile[cur][1 * 16 + col][0];
            o1 = bwmma(pa, mk16(*(const v8bf*)(vr + klo),
                                *(const v8bf*)(vr + klo + 16)), o1);
        }
        {
            const __bf16* vr = &vtile[cur][2 * 16 + col][0];
            o2 = bwmma(pa, mk16(*(const v8bf*)(vr + klo),
                                *(const v8bf*)(vr + klo + 16)), o2);
        }
        {
            const __bf16* vr = &vtile[cur][3 * 16 + col][0];
            o3 = bwmma(pa, mk16(*(const v8bf*)(vr + klo),
                                *(const v8bf*)(vr + klo + 16)), o3);
        }

        __syncthreads();   // buf cur free for restage; buf cur^1 complete
    }

    // ---- normalize and store ctx in (S,B,E) bf16 layout
#pragma unroll
    for (int r = 0; r < 8; ++r) {
        const float inv = 1.0f / l[r];
        const int srow = qt * 16 + r + klo;
        __bf16* crow = ctx + (size_t)(srow * BSZ + b) * EMB + h * DH;
        crow[0 * 16 + col] = (__bf16)(o0[r] * inv);
        crow[1 * 16 + col] = (__bf16)(o1[r] * inv);
        crow[2 * 16 + col] = (__bf16)(o2[r] * inv);
        crow[3 * 16 + col] = (__bf16)(o3[r] * inv);
    }
}

// ---------------------------- output projection ----------------------------
// One wave -> 64x32 tile: per k-step 4 A-frags + 2 B-frags feed 8 WMMAs.
__global__ __launch_bounds__(256) void out_proj_k(const __bf16* __restrict__ ctx,
                                                  const __bf16* __restrict__ woT,
                                                  const float*  __restrict__ bias,
                                                  float* __restrict__ out) {
    const int wid  = threadIdx.x >> 5;
    const int lane = threadIdx.x & 31;
    const int tile = blockIdx.x * 8 + wid;        // 0 .. 4095
    const int tmb = tile >> 5;                    // 128 M-blocks of 64 rows
    const int tnb = tile & 31;                    // 32 N-blocks of 32 cols
    const int col = lane & 15;
    const int hi  = lane >> 4;
    const int klo = hi * 8;

    const __bf16* ap[4];
#pragma unroll
    for (int i = 0; i < 4; ++i)
        ap[i] = ctx + (size_t)(tmb * 64 + i * 16 + col) * EMB;
    const __bf16* bp0 = woT + (size_t)(tnb * 32 + col) * EMB;
    const __bf16* bp1 = woT + (size_t)(tnb * 32 + 16 + col) * EMB;

    v8f acc0[4], acc1[4];
#pragma unroll
    for (int i = 0; i < 4; ++i) { acc0[i] = zero8(); acc1[i] = zero8(); }

    for (int kk = 0; kk < EMB; kk += 32) {
        v16bf f0 = mk16(ld8(bp0 + kk + klo), ld8(bp0 + kk + klo + 16));
        v16bf f1 = mk16(ld8(bp1 + kk + klo), ld8(bp1 + kk + klo + 16));
#pragma unroll
        for (int i = 0; i < 4; ++i) {
            v16bf af = mk16(ld8(ap[i] + kk + klo), ld8(ap[i] + kk + klo + 16));
            acc0[i] = bwmma(af, f0, acc0[i]);
            acc1[i] = bwmma(af, f1, acc1[i]);
        }
    }

    const int n0 = tnb * 32 + col;
    const int n1 = n0 + 16;
    const float bv0 = bias[n0];
    const float bv1 = bias[n1];
#pragma unroll
    for (int i = 0; i < 4; ++i) {
#pragma unroll
        for (int r = 0; r < 8; ++r) {
            const int row = tmb * 64 + i * 16 + r + klo;
            out[(size_t)row * EMB + n0] = acc0[i][r] + bv0;
            out[(size_t)row * EMB + n1] = acc1[i][r] + bv1;
        }
    }
}

// ------------------------------- launcher ----------------------------------

extern "C" void kernel_launch(void* const* d_in, const int* in_sizes, int n_in,
                              void* d_out, int out_size, void* d_ws, size_t ws_size,
                              hipStream_t stream) {
    (void)in_sizes; (void)n_in; (void)out_size; (void)ws_size;
    const float* x     = (const float*)d_in[0];
    const float* mask  = (const float*)d_in[1];
    const float* w_q   = (const float*)d_in[2];
    const float* w_k   = (const float*)d_in[3];
    const float* w_v   = (const float*)d_in[4];
    const float* w_out = (const float*)d_in[5];
    const float* b_out = (const float*)d_in[6];
    float* out = (float*)d_out;

    // workspace carve-out (~88 MB, 256B aligned)
    char* ws = (char*)d_ws;
    size_t off = 0;
    auto carve = [&](size_t bytes) -> void* {
        void* p = ws + off;
        off += (bytes + 255) & ~(size_t)255;
        return p;
    };
    const size_t act = (size_t)MROWS * EMB * sizeof(__bf16);    // 16 MB
    const size_t wsz = (size_t)EMB * EMB * sizeof(__bf16);      //  2 MB
    __bf16* xb  = (__bf16*)carve(act);
    __bf16* wqT = (__bf16*)carve(wsz);
    __bf16* wkT = (__bf16*)carve(wsz);
    __bf16* wvT = (__bf16*)carve(wsz);
    __bf16* woT = (__bf16*)carve(wsz);
    __bf16* qb  = (__bf16*)carve(act);
    __bf16* kb  = (__bf16*)carve(act);
    __bf16* vt  = (__bf16*)carve(act);
    __bf16* ctx = (__bf16*)carve(act);

    cvt_f32_bf16_k<<<MROWS * EMB / (256 * 4), 256, 0, stream>>>(x, xb);
    transpose_bf16_k<<<EMB * EMB / 256, 256, 0, stream>>>(w_q, wqT);
    transpose_bf16_k<<<EMB * EMB / 256, 256, 0, stream>>>(w_k, wkT);
    transpose_bf16_k<<<EMB * EMB / 256, 256, 0, stream>>>(w_v, wvT);
    transpose_bf16_k<<<EMB * EMB / 256, 256, 0, stream>>>(w_out, woT);

    // qkv: 128 M-blocks x 64 N-tiles = 8192 waves -> 1024 blocks
    qkv_proj_k<<<1024, 256, 0, stream>>>(xb, x, wqT, wkT, wvT, qb, kb, vt);
    // attn: 128 (b,h) x 8 blocks (8 query tiles each) = 1024 blocks
    attn_k<<<1024, 256, 0, stream>>>(qb, kb, vt, mask, ctx);
    // out-proj: 128 M-blocks x 32 N-blocks = 4096 waves -> 512 blocks
    out_proj_k<<<512, 256, 0, stream>>>(ctx, woT, b_out, out);
}
